// QuantLinearMLX_44959717655209
// MI455X (gfx1250) — matrix operand order repaired and checked by reading, not compiled, not run on hardware
//
#include <hip/hip_runtime.h>
#include <hip/hip_bf16.h>

// ---- problem constants (match reference) ----
#define IN_F     4096
#define OUT_F    4096
#define PACKED   (IN_F / 8)     // 512 int32 words per output row
#define NGROUPS  (IN_F / 128)   // 32 quant groups per output row

// ---- tiling ----
#define BM  128
#define BN  128
#define BK  64                  // one K-step stays inside one quant group
#define LDA 72                  // padded LDS stride (elements) for A tile
#define LDB 72                  // padded LDS stride (elements) for W tile
#define NT  (IN_F / BK)         // 64 K-tiles

typedef __attribute__((ext_vector_type(16))) __bf16 v16bf;
typedef __attribute__((ext_vector_type(8)))  __bf16 v8bf;
typedef __attribute__((ext_vector_type(4)))  __bf16 v4bf;
typedef __attribute__((ext_vector_type(8)))  float  v8f;
typedef __attribute__((ext_vector_type(4)))  float  v4f;

__global__ __launch_bounds__(256)
void qgemm_wmma_bf16(const float* __restrict__ x,
                     const int*   __restrict__ wq,
                     const float* __restrict__ scales,
                     const float* __restrict__ biases,
                     const float* __restrict__ bias,
                     float*       __restrict__ out)
{
    // double-buffered tiles: 2 * (9216 + 9216) * 2B = 72 KB LDS
    __shared__ __bf16 sA[2][BM * LDA];   // x tile, [m][k], bf16
    __shared__ __bf16 sB[2][BN * LDB];   // dequantized W tile, [n][k], bf16

    const int tid  = threadIdx.x;
    const int lane = tid & 31;
    const int wid  = tid >> 5;        // 0..7
    const int wrow = wid & 3;         // wave's 32-row slab within BM
    const int wcol = wid >> 2;        // wave's 64-col slab within BN

    const int m_base = blockIdx.y * BM;
    const int n_base = blockIdx.x * BN;

    const int lhi = lane >> 4;        // 0 | 1 (half-wave)
    const int llo = lane & 15;

    // staging thread roles
    const int a_c4 = tid & 15;        // float4 column within K-step (16 per row)
    const int a_r0 = tid >> 4;        // 0..15  (strided by 16 over 128 rows)
    const int w_w  = tid & 7;         // packed word within K-step
    const int w_r0 = tid >> 3;        // 0..31  (strided by 32 over 128 rows)

    v8f acc[2][4];
#pragma unroll
    for (int mt = 0; mt < 2; ++mt)
#pragma unroll
        for (int nt = 0; nt < 4; ++nt)
            acc[mt][nt] = (v8f)(0.0f);

    // staging registers (live across compute to pipeline global latency)
    v4f      ra[8];
    unsigned rw[4];
    float    rs[4], rb[4];

    auto fetch = [&](int k0) {
#pragma unroll
        for (int i = 0; i < 8; ++i)
            ra[i] = *(const v4f*)(x + (size_t)(m_base + a_r0 + i * 16) * IN_F
                                    + k0 + a_c4 * 4);
        const int g = k0 >> 7;        // quant group (uniform over a BK=64 tile)
#pragma unroll
        for (int i = 0; i < 4; ++i) {
            const int n = n_base + w_r0 + i * 32;
            rw[i] = (unsigned)wq[(size_t)n * PACKED + (k0 >> 3) + w_w];
            rs[i] = scales[(size_t)n * NGROUPS + g];
            rb[i] = biases[(size_t)n * NGROUPS + g];
        }
    };

    auto stage = [&](int buf) {
#pragma unroll
        for (int i = 0; i < 8; ++i) {
            v4bf pk;
            pk[0] = (__bf16)ra[i].x; pk[1] = (__bf16)ra[i].y;
            pk[2] = (__bf16)ra[i].z; pk[3] = (__bf16)ra[i].w;
            *(v4bf*)&sA[buf][(a_r0 + i * 16) * LDA + a_c4 * 4] = pk;
        }
#pragma unroll
        for (int i = 0; i < 4; ++i) {
            v8bf pk;
#pragma unroll
            for (int j = 0; j < 8; ++j) {
                const float q = (float)((rw[i] >> (4 * j)) & 15u);
                pk[j] = (__bf16)(rs[i] * q + rb[i]);
            }
            *(v8bf*)&sB[buf][(w_r0 + i * 32) * LDB + w_w * 8] = pk;
        }
    };

    // ---- prologue: stage tile 0 ----
    fetch(0);
    stage(0);
    __syncthreads();

    // ---- pipelined main loop ----
    for (int it = 0; it < NT; ++it) {
        const int cur = it & 1;

        // issue next tile's global loads early; latency hides under WMMA below
        if (it + 1 < NT) fetch((it + 1) * BK);

        // prefetch two tiles ahead (global_prefetch_b8)
        if (it + 2 < NT) {
            const int kp = (it + 2) * BK;
            __builtin_prefetch(wq + (size_t)(n_base + w_r0) * PACKED + (kp >> 3) + w_w, 0, 3);
            __builtin_prefetch(x  + (size_t)(m_base + a_r0) * IN_F + kp + a_c4 * 4,   0, 3);
        }

        // compute current tile: 2 K32 slices x 8 WMMAs
#pragma unroll
        for (int kk = 0; kk < BK; kk += 32) {
            // A fragments (16x32 bf16): lanes 0-15 -> K [0,16); 16-31 -> K [16,32)
            v16bf af[2];
#pragma unroll
            for (int mt = 0; mt < 2; ++mt) {
                const __bf16* p = &sA[cur][(wrow * 32 + mt * 16 + llo) * LDA
                                           + kk + lhi * 16];
                v8bf a0 = *(const v8bf*)p;
                v8bf a1 = *(const v8bf*)(p + 8);
                af[mt] = __builtin_shufflevector(a0, a1,
                    0,1,2,3,4,5,6,7,8,9,10,11,12,13,14,15);
            }
            // B fragments (32x16 bf16): col = llo, 16 contiguous K per half-wave
            v16bf bf4[4];
#pragma unroll
            for (int nt = 0; nt < 4; ++nt) {
                const __bf16* p = &sB[cur][(wcol * 64 + nt * 16 + llo) * LDB
                                           + kk + lhi * 16];
                v8bf b0 = *(const v8bf*)p;
                v8bf b1 = *(const v8bf*)(p + 8);
                bf4[nt] = __builtin_shufflevector(b0, b1,
                    0,1,2,3,4,5,6,7,8,9,10,11,12,13,14,15);
            }
#pragma unroll
            for (int nt = 0; nt < 4; ++nt)
#pragma unroll
                for (int mt = 0; mt < 2; ++mt)
                    acc[mt][nt] = __builtin_amdgcn_wmma_f32_16x16x32_bf16(
                        false, af[mt], false, bf4[nt],
                        (short)0, acc[mt][nt], false, false);
        }

        // convert + store next tile into the other buffer
        if (it + 1 < NT) stage(cur ^ 1);

        __syncthreads();
    }

    // ---- epilogue: add bias[n], store fp32 ----
#pragma unroll
    for (int mt = 0; mt < 2; ++mt) {
#pragma unroll
        for (int nt = 0; nt < 4; ++nt) {
            const int n    = n_base + wcol * 64 + nt * 16 + llo;
            const float bv = bias[n];
            const int mrow = m_base + wrow * 32 + mt * 16 + lhi * 8;
#pragma unroll
            for (int v = 0; v < 8; ++v) {
                out[(size_t)(mrow + v) * OUT_F + n] = acc[mt][nt][v] + bv;
            }
        }
    }
}

extern "C" void kernel_launch(void* const* d_in, const int* in_sizes, int n_in,
                              void* d_out, int out_size, void* d_ws, size_t ws_size,
                              hipStream_t stream) {
    const float* x      = (const float*)d_in[0];
    const int*   wq     = (const int*)  d_in[1];
    const float* scales = (const float*)d_in[2];
    const float* biases = (const float*)d_in[3];
    const float* bias   = (const float*)d_in[4];
    float*       out    = (float*)d_out;

    const int M = in_sizes[0] / IN_F;     // 4*2048 = 8192
    dim3 grid(OUT_F / BN, M / BM);        // 32 x 64 = 2048 workgroups
    qgemm_wmma_bf16<<<grid, 256, 0, stream>>>(x, wq, scales, biases, bias, out);
}
